// Convolution_FKAConv_21345987461192
// MI455X (gfx1250) — matrix operand order repaired and checked by reading, not compiled, not run on hardware
//
#include <hip/hip_runtime.h>
#include <math.h>

// FKAConv fused kernel for MI455X (gfx1250, wave32, WMMA).
// B=4, CIN=COUT=64, N=32768, S=8192, K=32 neighbors, KS=16, DIM=3.
// wave = support point for the MLP (lane = neighbor); fc2/fc3, the
// neighbor-feature GEMM and the output GEMM all run on v_wmma_f32_16x16x32_f16.
// Neighbor-max reductions run packed (v_pk_max_num_f16) to halve bpermute/VALU.

#define B_    4
#define CIN   64
#define COUT  64
#define NPTS  32768
#define S_    8192
#define KNB   32
#define KSZ   16
#define DIM_  3
#define STILE 16          // support points per block
#define THREADS 512       // 16 waves of 32
#define KW    (CIN * KSZ) // 1024 = GEMM2 K extent

typedef __attribute__((ext_vector_type(16))) _Float16 v16h;
typedef __attribute__((ext_vector_type(2)))  _Float16 h2;
typedef __attribute__((ext_vector_type(8)))  float    v8f;
typedef __attribute__((ext_vector_type(4)))  float    v4f;
typedef __attribute__((ext_vector_type(4)))  int      v4i;

__device__ __forceinline__ int h2i(h2 v) { return __builtin_bit_cast(int, v); }
__device__ __forceinline__ h2  i2h(int v) { return __builtin_bit_cast(h2, v); }

union AFrag { v4i i2[2]; v16h h; };

// K index held by (half h, lane) in a 16-bit 16x32 A fragment:
// lanes 0-15: h=0..7 -> K=h, h=8..15 -> K=h+8 ; lanes 16-31: +8.
__device__ __forceinline__ int kmapA(int h, int lane) {
    return h + (h & 8) + ((lane & 16) >> 1);
}

// Build an f16 A fragment from a row-major f32 matrix row (one 16-float run at k offset 16*(lane/16)).
__device__ __forceinline__ v16h load_a_frag_f32(const float* __restrict__ base,
                                                int row, int ldm, int kbase, int lane) {
    const float* wp = base + row * ldm + kbase + (lane & 16);
    const v4f w0 = *(const v4f*)(wp);
    const v4f w1 = *(const v4f*)(wp + 4);
    const v4f w2 = *(const v4f*)(wp + 8);
    const v4f w3 = *(const v4f*)(wp + 12);
    v16h a;
    #pragma unroll
    for (int j = 0; j < 4; ++j) {
        a[j]      = (_Float16)w0[j];
        a[4 + j]  = (_Float16)w1[j];
        a[8 + j]  = (_Float16)w2[j];
        a[12 + j] = (_Float16)w3[j];
    }
    return a;
}

// Prep: wcvT[o][k'*64 + c] = (f16) wcv[o][c][k']  (GEMM2 K-order transpose).
__global__ void wcv_prep_kernel(const float* __restrict__ wcv, _Float16* __restrict__ wcvT) {
    const int i = blockIdx.x * 256 + threadIdx.x;     // 0 .. 65535
    const int o = i >> 10, kidx = i & 1023;
    const int kp = kidx >> 6, c = kidx & 63;
    wcvT[i] = (_Float16)wcv[o * KW + c * KSZ + kp];
}

template <bool USE_WS>
__global__ __launch_bounds__(THREADS)
void fkaconv_kernel(const float* __restrict__ x,
                    const float* __restrict__ pos,
                    const float* __restrict__ sup,
                    const long long* __restrict__ nbr,
                    const float* __restrict__ radius,
                    const float* __restrict__ wf1,
                    const float* __restrict__ wf2,
                    const float* __restrict__ wf3,
                    const float* __restrict__ wcv,
                    const _Float16* __restrict__ wcvT,
                    float* __restrict__ out)
{
    __shared__ __align__(16) int s_idx[STILE][KNB];
    __shared__ __align__(32) _Float16 s_mat[STILE][KSZ][KNB];   // [s][k'][n]
    // Dual-purpose per-wave buffer (phases disjoint, wave-local until GEMM2):
    //   MLP phase : 32x32 activation matrix, layout [n][m] f16
    //   GEMM phase: feats; K-order is k'*64+c (USE_WS) or c*16+k' (fallback)
    __shared__ __align__(32) _Float16 s_big[STILE][KW];

    const int tid  = threadIdx.x;
    const int wave = tid >> 5;
    const int lane = tid & 31;
    const int b    = blockIdx.x / (S_ / STILE);
    const int s0   = (blockIdx.x % (S_ / STILE)) * STILE;
    const int l15  = lane & 15;
    const int hi   = (lane & 16) >> 1;   // 0 or 8  (A/D row-half offset)
    const int hb   = lane & 16;          // 0 or 16 (B row-half offset)

    _Float16* vrow = &s_big[wave][0];    // MLP activation scratch [n*32 + m]

    // ================= Stage 1: geometry MLP (wave = support point) =========
    float dw, dw0, dw1;
    {
        const int s = s0 + wave;
        const long long raw = nbr[((long long)(b * S_ + s)) * KNB + lane];
        const bool m  = raw > -1;
        const int idx = m ? (int)raw : 0;
        s_idx[wave][lane] = idx;

        float p[DIM_];
        #pragma unroll
        for (int d = 0; d < DIM_; ++d) {
            const float pg = pos[(b * DIM_ + d) * NPTS + idx];
            p[d] = pg - sup[(b * DIM_ + d) * S_ + s];
        }
        const float maskp = isinf(p[0]) ? 0.0f : 1.0f;
        const float inv_r = 1.0f / radius[0];
        #pragma unroll
        for (int d = 0; d < DIM_; ++d) {
            float v = p[d];
            if (!isfinite(v)) v = 0.0f;              // nan_to_num
            p[d] = v * inv_r;
        }
        const unsigned long long bal = __ballot(m ? 1 : 0);
        dw = (m ? 1.0f : 0.0f) / fmaxf(sqrtf((float)__popcll(bal)), 1e-12f) * maskp;

        // fc1 (3->16): tiny, VALU; weights wave-uniform -> scalar loads.
        h2 r0p[8];
        #pragma unroll
        for (int j = 0; j < 8; ++j) {
            float acc0 = 0.0f, acc1 = 0.0f;
            #pragma unroll
            for (int d = 0; d < DIM_; ++d) {
                acc0 += wf1[(2 * j) * DIM_ + d] * p[d];
                acc1 += wf1[(2 * j + 1) * DIM_ + d] * p[d];
            }
            h2 t; t[0] = (_Float16)(fmaxf(acc0, 0.0f) * dw);
                  t[1] = (_Float16)(fmaxf(acc1, 0.0f) * dw);
            r0p[j] = t;
        }
        // mp1: packed max over 32 neighbor lanes
        h2 m1[8];
        #pragma unroll
        for (int j = 0; j < 8; ++j) m1[j] = r0p[j];
        #pragma unroll
        for (int off = 16; off > 0; off >>= 1) {
            #pragma unroll
            for (int j = 0; j < 8; ++j)
                m1[j] = __builtin_elementwise_max(m1[j], i2h(__shfl_xor(h2i(m1[j]), off, 32)));
        }
        {
            int* rowi = (int*)&vrow[lane * 32];
            v4i wa = { h2i(r0p[0]), h2i(r0p[1]), h2i(r0p[2]), h2i(r0p[3]) };
            v4i wb = { h2i(r0p[4]), h2i(r0p[5]), h2i(r0p[6]), h2i(r0p[7]) };
            v4i wc = { h2i(m1[0]),  h2i(m1[1]),  h2i(m1[2]),  h2i(m1[3])  };
            v4i wd = { h2i(m1[4]),  h2i(m1[5]),  h2i(m1[6]),  h2i(m1[7])  };
            *(v4i*)(rowi + 0)  = wa;  *(v4i*)(rowi + 4)  = wb;
            *(v4i*)(rowi + 8)  = wc;  *(v4i*)(rowi + 12) = wd;
        }
        __builtin_amdgcn_wave_barrier();

        // dw for the D-fragment columns this lane will own:
        dw0 = __shfl(dw, l15, 32);
        dw1 = __shfl(dw, 16 + l15, 32);

        // ---- fc2 (WMMA): mat2[16x32] = wf2[16x32] @ V[32x32] ----
        {
            const v16h af = load_a_frag_f32(wf2, l15, 2 * KSZ, 0, lane);
            const v16h b0 = *(const v16h*)&vrow[l15 * 32 + hb];
            const v16h b1 = *(const v16h*)&vrow[(16 + l15) * 32 + hb];
            v8f d0 = {}, d1 = {};
            d0 = __builtin_amdgcn_wmma_f32_16x16x32_f16(false, af, false, b0, (short)0, d0, false, false);
            d1 = __builtin_amdgcn_wmma_f32_16x16x32_f16(false, af, false, b1, (short)0, d1, false, false);
            h2 q0[4], q1[4], pm[4];
            #pragma unroll
            for (int j = 0; j < 4; ++j) {
                h2 t0, t1;
                t0[0] = (_Float16)(fmaxf(d0[2 * j], 0.0f) * dw0);
                t0[1] = (_Float16)(fmaxf(d0[2 * j + 1], 0.0f) * dw0);
                t1[0] = (_Float16)(fmaxf(d1[2 * j], 0.0f) * dw1);
                t1[1] = (_Float16)(fmaxf(d1[2 * j + 1], 0.0f) * dw1);
                q0[j] = t0; q1[j] = t1;
                pm[j] = __builtin_elementwise_max(t0, t1);
            }
            #pragma unroll
            for (int off = 8; off > 0; off >>= 1) {   // reduce within 16-lane half
                #pragma unroll
                for (int j = 0; j < 4; ++j)
                    pm[j] = __builtin_elementwise_max(pm[j], i2h(__shfl_xor(h2i(pm[j]), off, 32)));
            }
            const v4i s0v = { h2i(q0[0]), h2i(q0[1]), h2i(q0[2]), h2i(q0[3]) };
            const v4i s1v = { h2i(q1[0]), h2i(q1[1]), h2i(q1[2]), h2i(q1[3]) };
            const v4i smv = { h2i(pm[0]), h2i(pm[1]), h2i(pm[2]), h2i(pm[3]) };
            __builtin_amdgcn_wave_barrier();
            *(v4i*)&vrow[l15 * 32 + hi]             = s0v;   // mat2, cols 0..15
            *(v4i*)&vrow[(16 + l15) * 32 + hi]      = s1v;   // mat2, cols 16..31
            *(v4i*)&vrow[l15 * 32 + 16 + hi]        = smv;   // mp2 broadcast
            *(v4i*)&vrow[(16 + l15) * 32 + 16 + hi] = smv;
            __builtin_amdgcn_wave_barrier();
        }

        // ---- fc3 (WMMA): mat3 -> s_mat[s][k'][n] ----
        {
            const v16h af = load_a_frag_f32(wf3, l15, 2 * KSZ, 0, lane);
            const v16h b0 = *(const v16h*)&vrow[l15 * 32 + hb];
            const v16h b1 = *(const v16h*)&vrow[(16 + l15) * 32 + hb];
            v8f d0 = {}, d1 = {};
            d0 = __builtin_amdgcn_wmma_f32_16x16x32_f16(false, af, false, b0, (short)0, d0, false, false);
            d1 = __builtin_amdgcn_wmma_f32_16x16x32_f16(false, af, false, b1, (short)0, d1, false, false);
            #pragma unroll
            for (int r = 0; r < 8; ++r) {
                s_mat[wave][hi + r][l15]      = (_Float16)(fmaxf(d0[r], 0.0f) * dw0);
                s_mat[wave][hi + r][16 + l15] = (_Float16)(fmaxf(d1[r], 0.0f) * dw1);
            }
        }
    }
    __syncthreads();

    // ========== Stage 2 (GEMM1, WMMA): feats[c][k'] = sum_n Xg[c][n]*mat[k'][n]
    {
        const int sl = wave;
        const v16h bfrag = *(const v16h*)&s_mat[sl][l15][hb];
        // contiguous K order: one 16-int run of s_idx at hb -> 4x ds_load_b128.
        const v4i g0 = *(const v4i*)&s_idx[sl][hb];
        const v4i g1 = *(const v4i*)&s_idx[sl][hb + 4];
        const v4i g2 = *(const v4i*)&s_idx[sl][hb + 8];
        const v4i g3 = *(const v4i*)&s_idx[sl][hb + 12];
        int gidx[16];
        #pragma unroll
        for (int j = 0; j < 4; ++j) {
            gidx[j] = g0[j]; gidx[4 + j] = g1[j];
            gidx[8 + j] = g2[j]; gidx[12 + j] = g3[j];
        }
        #pragma unroll
        for (int mt = 0; mt < 4; ++mt) {
            const int c = mt * 16 + l15;
            v16h afrag;
            #pragma unroll
            for (int h = 0; h < 16; ++h)
                afrag[h] = (_Float16)x[(b * CIN + c) * NPTS + gidx[h]];
            v8f cacc = {};
            cacc = __builtin_amdgcn_wmma_f32_16x16x32_f16(
                false, afrag, false, bfrag, (short)0, cacc, false, false);
            if (USE_WS) {
                // feats[k'*64 + c]: lane's 8 rows are consecutive -> one 16B store
                h2 f0, f1, f2, f3;
                f0[0] = (_Float16)cacc[0]; f0[1] = (_Float16)cacc[1];
                f1[0] = (_Float16)cacc[2]; f1[1] = (_Float16)cacc[3];
                f2[0] = (_Float16)cacc[4]; f2[1] = (_Float16)cacc[5];
                f3[0] = (_Float16)cacc[6]; f3[1] = (_Float16)cacc[7];
                const v4i fv = { h2i(f0), h2i(f1), h2i(f2), h2i(f3) };
                *(v4i*)&s_big[sl][l15 * 64 + mt * 16 + hi] = fv;
            } else {
                #pragma unroll
                for (int r = 0; r < 8; ++r)
                    s_big[sl][(mt * 16 + hi + r) * KSZ + l15] = (_Float16)cacc[r];
            }
        }
    }
    __syncthreads();

    // ========== Stage 3 (GEMM2, WMMA): out[o][s] = wcv[o][:] . feats[s][:]
    if (wave < 4) {
        const int mt = wave;
        v8f acc = {};
        for (int kk = 0; kk < KW / 32; ++kk) {
            AFrag af;
            if (USE_WS) {   // pre-transposed f16 weights: two 16B loads, no cvt
                const _Float16* wa = wcvT + (size_t)(mt * 16 + l15) * KW + kk * 32 + hb;
                af.i2[0] = *(const v4i*)(wa);
                af.i2[1] = *(const v4i*)(wa + 8);
            } else {
                af.h = load_a_frag_f32(wcv, mt * 16 + l15, KW, kk * 32, lane);
            }
            const v16h bfrag = *(const v16h*)&s_big[l15][kk * 32 + hb];
            acc = __builtin_amdgcn_wmma_f32_16x16x32_f16(
                false, af.h, false, bfrag, (short)0, acc, false, false);
        }
        const int sg = s0 + l15;
        const bool supinf = isinf(sup[(b * DIM_ + 0) * S_ + sg]);
        #pragma unroll
        for (int r = 0; r < 8; ++r) {
            const int oo = mt * 16 + hi + r;
            float v = acc[r];
            if (supinf) v = INFINITY;
            out[((size_t)(b * COUT + oo)) * S_ + sg] = v;
        }
    }
}

extern "C" void kernel_launch(void* const* d_in, const int* in_sizes, int n_in,
                              void* d_out, int out_size, void* d_ws, size_t ws_size,
                              hipStream_t stream) {
    (void)in_sizes; (void)n_in; (void)out_size;
    const float*     x   = (const float*)d_in[0];
    const float*     pos = (const float*)d_in[1];
    const float*     sup = (const float*)d_in[2];
    const long long* nbr = (const long long*)d_in[3];
    const float*     rad = (const float*)d_in[4];
    const float*     wf1 = (const float*)d_in[5];
    const float*     wf2 = (const float*)d_in[6];
    const float*     wf3 = (const float*)d_in[7];
    const float*     wcv = (const float*)d_in[8];
    float*           out = (float*)d_out;

    const size_t ws_needed = (size_t)COUT * KW * sizeof(_Float16);   // 128 KB
    dim3 grid(B_ * (S_ / STILE));   // 2048 blocks
    dim3 block(THREADS);            // 16 waves

    if (ws_size >= ws_needed && d_ws != nullptr) {
        _Float16* wcvT = (_Float16*)d_ws;
        wcv_prep_kernel<<<dim3((COUT * KW) / 256), dim3(256), 0, stream>>>(wcv, wcvT);
        fkaconv_kernel<true><<<grid, block, 0, stream>>>(
            x, pos, sup, nbr, rad, wf1, wf2, wf3, wcv, wcvT, out);
    } else {
        fkaconv_kernel<false><<<grid, block, 0, stream>>>(
            x, pos, sup, nbr, rad, wf1, wf2, wf3, wcv, nullptr, out);
    }
}
